// VectorQuantizer_69750268887140
// MI455X (gfx1250) — matrix operand order repaired and checked by reading, not compiled, not run on hardware
//
#include <hip/hip_runtime.h>
#include <hip/hip_bf16.h>

#define M_ROWS 32768
#define K_CODES 8192
#define D_DIM 256
#define BETA_F 0.05f
#define EPS_N 1e-8f
#define N_WAVES_TOTAL (M_ROWS / 16)   // 2048 search waves
#define N_TILES (K_CODES / 16)        // 512 K-tiles
#define LDS_STRIDE 528                // padded LDS row stride (bytes)
#define LDS_BUF (16 * LDS_STRIDE)     // one staged tile: 16 codes x 256 bf16 (+pad)

typedef __attribute__((ext_vector_type(16))) __bf16 v16bf;
typedef __attribute__((ext_vector_type(8)))  __bf16 v8bf;
typedef __attribute__((ext_vector_type(8)))  float  v8f;
typedef __attribute__((ext_vector_type(4)))  int    v4i;

// Address-space-qualified 128-bit payload types for the async-copy builtin:
// param 1 is v4i in AS(1) (printed as "__device__" by clang), param 2 is LDS.
typedef __attribute__((address_space(1))) v4i g_v4i;
typedef __attribute__((address_space(3))) v4i l_v4i;

// ---------------------------------------------------------------------------
// gfx1250 async global->LDS copy (ASYNCcnt path), with synchronous fallback.
// ---------------------------------------------------------------------------
__device__ __forceinline__ void async_copy16(const void* g, void* l) {
#if __has_builtin(__builtin_amdgcn_global_load_async_to_lds_b128)
    __builtin_amdgcn_global_load_async_to_lds_b128(
        (g_v4i*)(v4i*)g, (l_v4i*)(v4i*)l, 0, 0);
#else
    *(float4*)l = *(const float4*)g;
#endif
}

__device__ __forceinline__ void async_wait0() {
#if __has_builtin(__builtin_amdgcn_s_wait_asynccnt)
    __builtin_amdgcn_s_wait_asynccnt(0);
#endif
}

// ---------------------------------------------------------------------------
// Kernel 1: E_n = normalize(E) rows, stored bf16 row-major [K, D] in ws.
// One wave per codebook row: 32 lanes x 8 elements = 256.
// ---------------------------------------------------------------------------
__global__ void __launch_bounds__(256)
vq_norm_E(const float* __restrict__ E, __bf16* __restrict__ En) {
    const int wave = (blockIdx.x * blockDim.x + threadIdx.x) >> 5;  // row id
    const int lane = threadIdx.x & 31;
    const float* row = E + (size_t)wave * D_DIM + lane * 8;
    float4 a = *(const float4*)(row);
    float4 b = *(const float4*)(row + 4);
    float s = a.x*a.x + a.y*a.y + a.z*a.z + a.w*a.w
            + b.x*b.x + b.y*b.y + b.z*b.z + b.w*b.w;
#pragma unroll
    for (int m = 1; m < 32; m <<= 1) s += __shfl_xor(s, m, 32);
    const float inv = 1.0f / fmaxf(sqrtf(s), EPS_N);
    v8bf o;
    o[0] = (__bf16)(a.x * inv); o[1] = (__bf16)(a.y * inv);
    o[2] = (__bf16)(a.z * inv); o[3] = (__bf16)(a.w * inv);
    o[4] = (__bf16)(b.x * inv); o[5] = (__bf16)(b.y * inv);
    o[6] = (__bf16)(b.z * inv); o[7] = (__bf16)(b.w * inv);
    *(v8bf*)(En + (size_t)wave * D_DIM + lane * 8) = o;
}

// Load one 32-byte B-fragment chunk from LDS (two 16B reads, 16B aligned).
__device__ __forceinline__ v16bf lds_frag(const char* p) {
    v8bf lo = *(const v8bf*)(p);
    v8bf hi = *(const v8bf*)(p + 16);
    v16bf b;
#pragma unroll
    for (int e = 0; e < 8; ++e) { b[e] = lo[e]; b[8 + e] = hi[e]; }
    return b;
}

// ---------------------------------------------------------------------------
// Kernel 2: WMMA cosine search. One wave owns 16 z rows. The 8 waves of a
// workgroup share double-buffered E_n tiles staged into LDS with async
// global->LDS copies; per tile each wave runs two independent 4-deep
// v_wmma_f32_16x16x32_bf16 accumulation chains over D=256.
// ---------------------------------------------------------------------------
__global__ void __launch_bounds__(256)
vq_search(const float* __restrict__ z, const float* __restrict__ E,
          const __bf16* __restrict__ En, float* __restrict__ out,
          float* __restrict__ partials) {
    __shared__ __align__(16) char lbuf[2][LDS_BUF];
    __shared__ int sidx[8][16];

    const int tid  = threadIdx.x;
    const int lane = tid & 31;
    const int w    = tid >> 5;
    const int gid  = blockIdx.x * 8 + w;          // 0 .. 2047
    const int m0   = gid * 16;                    // first of 16 rows
    const int h    = lane >> 4;                   // lane half
    const int r    = lane & 15;                   // lane-in-half

    // Per-thread staging assignment: 16 threads per tile row, 32 B each.
    const int srow = tid >> 4;                    // 0..15 (code within tile)
    const int sseg = tid & 15;                    // 0..15 (32B segment)

    // --- Load A fragments: raw z rows in bf16, row = lane % 16, contiguous K order.
    //   elems 0..15 -> K = 32c + 16h + e (contiguous, matches B)
    v16bf afrag[8];
    float sumsq = 0.0f;
    const float* zrow = z + (size_t)(m0 + r) * D_DIM;
#pragma unroll
    for (int c = 0; c < 8; ++c) {
        const int k0 = c * 32;
        const float* p1 = zrow + k0 + h * 16;
        const float* p2 = zrow + k0 + h * 16 + 8;
        float4 x0 = *(const float4*)(p1);
        float4 x1 = *(const float4*)(p1 + 4);
        float4 x2 = *(const float4*)(p2);
        float4 x3 = *(const float4*)(p2 + 4);
        sumsq += x0.x*x0.x + x0.y*x0.y + x0.z*x0.z + x0.w*x0.w;
        sumsq += x1.x*x1.x + x1.y*x1.y + x1.z*x1.z + x1.w*x1.w;
        sumsq += x2.x*x2.x + x2.y*x2.y + x2.z*x2.z + x2.w*x2.w;
        sumsq += x3.x*x3.x + x3.y*x3.y + x3.z*x3.z + x3.w*x3.w;
        v16bf a;
        a[0]  = (__bf16)x0.x; a[1]  = (__bf16)x0.y; a[2]  = (__bf16)x0.z; a[3]  = (__bf16)x0.w;
        a[4]  = (__bf16)x1.x; a[5]  = (__bf16)x1.y; a[6]  = (__bf16)x1.z; a[7]  = (__bf16)x1.w;
        a[8]  = (__bf16)x2.x; a[9]  = (__bf16)x2.y; a[10] = (__bf16)x2.z; a[11] = (__bf16)x2.w;
        a[12] = (__bf16)x3.x; a[13] = (__bf16)x3.y; a[14] = (__bf16)x3.z; a[15] = (__bf16)x3.w;
        afrag[c] = a;
    }
    sumsq += __shfl_xor(sumsq, 16, 32);
    const float inv_n = 1.0f / fmaxf(sqrtf(sumsq), EPS_N);

    float best_v[8];
    int   best_i[8];
#pragma unroll
    for (int i = 0; i < 8; ++i) { best_v[i] = -3.4e38f; best_i[i] = 0; }

    // --- Stage tile 0 into buffer 0 (async, ASYNCcnt-tracked).
    {
        const char* g = (const char*)(En + (size_t)srow * D_DIM) + sseg * 32;
        char* l = lbuf[0] + srow * LDS_STRIDE + sseg * 32;
        async_copy16(g, l);
        async_copy16(g + 16, l + 16);
    }
    async_wait0();
    __syncthreads();

#pragma unroll 1
    for (int t = 0; t < N_TILES; ++t) {
        // Prefetch next tile into the other buffer while computing this one.
        if (t + 1 < N_TILES) {
            const char* g = (const char*)(En + (size_t)((t + 1) * 16 + srow) * D_DIM)
                            + sseg * 32;
            char* l = lbuf[(t + 1) & 1] + srow * LDS_STRIDE + sseg * 32;
            async_copy16(g, l);
            async_copy16(g + 16, l + 16);
        }

        // B fragments from LDS: lane = tile column r, 32 contiguous K per chunk.
        const char* base = lbuf[t & 1] + r * LDS_STRIDE + h * 32;
        v8f acc0 = {}, acc1 = {};
#pragma unroll
        for (int c = 0; c < 4; ++c) {
            v16bf b = lds_frag(base + c * 64);
            acc0 = __builtin_amdgcn_wmma_f32_16x16x32_bf16(
                false, afrag[c], false, b, (short)0, acc0, false, false);
        }
#pragma unroll
        for (int c = 4; c < 8; ++c) {
            v16bf b = lds_frag(base + c * 64);
            acc1 = __builtin_amdgcn_wmma_f32_16x16x32_bf16(
                false, afrag[c], false, b, (short)0, acc1, false, false);
        }

        const int cand = t * 16 + r;
#pragma unroll
        for (int i = 0; i < 8; ++i) {
            float v = acc0[i] + acc1[i];      // row = i + 8h, col = cand
            bool gt = v > best_v[i];          // strict '>' keeps first index
            best_i[i] = gt ? cand : best_i[i];
            best_v[i] = gt ? v    : best_v[i];
        }

        async_wait0();       // next tile fully landed in LDS
        __syncthreads();     // everyone done reading current buffer
    }

    // --- Butterfly argmax over the 16 column-lanes of each half.
#pragma unroll
    for (int m = 1; m < 16; m <<= 1) {
#pragma unroll
        for (int i = 0; i < 8; ++i) {
            float ov = __shfl_xor(best_v[i], m, 32);
            int   oi = __shfl_xor(best_i[i], m, 32);
            bool take = (ov > best_v[i]) || (ov == best_v[i] && oi < best_i[i]);
            best_v[i] = take ? ov : best_v[i];
            best_i[i] = take ? oi : best_i[i];
        }
    }

    // --- Loss partial: sum over rows of (2 - 2*cos); cos = best_raw / ||z||.
    float halfsum = 0.0f;
#pragma unroll
    for (int i = 0; i < 8; ++i) {
        float invj = __shfl(inv_n, h * 8 + i, 32);   // row (h*8+i)'s 1/||z||
        halfsum += 2.0f - 2.0f * (best_v[i] * invj);
    }
    float wavesum = halfsum + __shfl_xor(halfsum, 16, 32);
    if (lane == 0) partials[gid] = wavesum;

    // --- Publish the 16 winning indices (one statically-indexed writer each).
#pragma unroll
    for (int i = 0; i < 8; ++i)
        if (r == i) sidx[w][h * 8 + i] = best_i[i];
    __syncthreads();

    float* out_zq  = out + 1;
    float* out_idx = out + 1 + (size_t)M_ROWS * D_DIM;
    if (lane < 16) out_idx[m0 + lane] = (float)sidx[w][lane];

    // --- Gather zq_st = E[idx] rows (straight-through forward value == zq).
#pragma unroll 1
    for (int j = 0; j < 16; ++j) {
        const float* src = E + (size_t)sidx[w][j] * D_DIM + lane * 8;
        float*       dst = out_zq + (size_t)(m0 + j) * D_DIM + lane * 8;
        float4 g0 = *(const float4*)(src);
        float4 g1 = *(const float4*)(src + 4);
        *(float4*)(dst)     = g0;
        *(float4*)(dst + 4) = g1;
    }
}

// ---------------------------------------------------------------------------
// Kernel 3: deterministic reduction of 2048 wave partials -> loss scalar.
// ---------------------------------------------------------------------------
__global__ void __launch_bounds__(256)
vq_reduce(const float* __restrict__ partials, float* __restrict__ out) {
    __shared__ float sbuf[256];
    float s = 0.0f;
    for (int i = threadIdx.x; i < N_WAVES_TOTAL; i += 256) s += partials[i];
    sbuf[threadIdx.x] = s;
    __syncthreads();
    for (int stride = 128; stride > 0; stride >>= 1) {
        if (threadIdx.x < (unsigned)stride) sbuf[threadIdx.x] += sbuf[threadIdx.x + stride];
        __syncthreads();
    }
    if (threadIdx.x == 0)
        out[0] = (1.0f + BETA_F) * sbuf[0] / (float)((size_t)M_ROWS * D_DIM);
}

// ---------------------------------------------------------------------------
extern "C" void kernel_launch(void* const* d_in, const int* in_sizes, int n_in,
                              void* d_out, int out_size, void* d_ws, size_t ws_size,
                              hipStream_t stream) {
    const float* z = (const float*)d_in[0];   // [M, D] fp32
    const float* E = (const float*)d_in[1];   // [K, D] fp32
    float* out = (float*)d_out;               // [1 + M*D + M]

    __bf16* En       = (__bf16*)d_ws;                                  // 4 MB
    float*  partials = (float*)((char*)d_ws +
                         (size_t)K_CODES * D_DIM * sizeof(__bf16));    // 8 KB

    vq_norm_E<<<K_CODES / 8, 256, 0, stream>>>(E, En);
    vq_search<<<M_ROWS / 128, 256, 0, stream>>>(z, E, En, out, partials);
    vq_reduce<<<1, 256, 0, stream>>>(partials, out);
}
